// RotarySelfAttention_42786464203256
// MI455X (gfx1250) — matrix operand order repaired and checked by reading, not compiled, not run on hardware
//
#include <hip/hip_runtime.h>

typedef unsigned short u16;
typedef __attribute__((ext_vector_type(8)))  u16    u16x8;
typedef __attribute__((ext_vector_type(16))) __bf16 bf16x16;
typedef __attribute__((ext_vector_type(8)))  float  f32x8;

#define LOG2E 1.4426950408889634f
#define NEGINF (-3.0e38f)

__device__ __forceinline__ u16 f32_to_bf16(float f) {
    unsigned x = __float_as_uint(f);
    x += 0x7fffu + ((x >> 16) & 1u);   // round-to-nearest-even
    return (u16)(x >> 16);
}

union FragBF { bf16x16 bf; u16x8 h[2]; };

// A-fragment (16x32 bf16): lane<16 holds row r, K[0..7]+K[16..23]; lane>=16 K[8..15]+K[24..31]
__device__ __forceinline__ bf16x16 frag_a(const u16* p, int ld, int lane) {
    int r = lane & 15, hi = lane >> 4;
    FragBF f;
    f.h[0] = *(const u16x8*)(p + r * ld + hi * 8);
    f.h[1] = *(const u16x8*)(p + r * ld + 16 + hi * 8);
    return f.bf;
}
// B-fragment (32x16 bf16) from N-major storage: lane holds column (lane&15), K[0..7]+K[16..23] (lane<16) / K[8..15]+K[24..31]
__device__ __forceinline__ bf16x16 frag_b(const u16* p, int ld, int lane) {
    int r = lane & 15, hi = lane >> 4;
    const u16* q = p + r * ld;
    FragBF f;
    f.h[0] = *(const u16x8*)(q + hi * 8);
    f.h[1] = *(const u16x8*)(q + 16 + hi * 8);
    return f.bf;
}
__device__ __forceinline__ f32x8 wmma_bf16(bf16x16 a, bf16x16 b, f32x8 c) {
    return __builtin_amdgcn_wmma_f32_16x16x32_bf16(false, a, false, b, (short)0, c,
                                                   false, false);
}

// CDNA5 direct global->LDS async DMA (16 bytes per lane), tracked by ASYNCcnt.
// GV mode: per-lane 64-bit global address in VGPR pair, per-lane LDS byte address in VDST.
__device__ __forceinline__ void async_cp16(void* lds_dst, const u16* gsrc) {
    unsigned lds = (unsigned)(unsigned long long)lds_dst;  // low 32 bits = LDS byte addr
    unsigned long long ga = (unsigned long long)gsrc;
    asm volatile("global_load_async_to_lds_b128 %0, %1, off"
                 :: "v"(lds), "v"(ga)
                 : "memory");
}
__device__ __forceinline__ void wait_async0() {
    asm volatile("s_wait_asynccnt 0x0" ::: "memory");
}

// ---------------------------------------------------------------- conversions
__global__ void f32_to_bf16_kernel(const float* __restrict__ in, u16* __restrict__ out, int n) {
    int i = blockIdx.x * 256 + threadIdx.x;
    if (i < n) out[i] = f32_to_bf16(in[i]);
}

// in: R x C (row-major f32) -> out: C x R (row-major bf16)
__global__ __launch_bounds__(256) void transpose_f32_bf16(const float* __restrict__ in,
                                                          u16* __restrict__ out, int R, int C) {
    __shared__ u16 tile[32][33];
    int c0 = blockIdx.x * 32, r0 = blockIdx.y * 32;
    int tx = threadIdx.x & 31, ty = threadIdx.x >> 5;   // ty in 0..7
#pragma unroll
    for (int i = 0; i < 32; i += 8)
        tile[ty + i][tx] = f32_to_bf16(in[(size_t)(r0 + ty + i) * C + c0 + tx]);
    __syncthreads();
#pragma unroll
    for (int i = 0; i < 32; i += 8)
        out[(size_t)(c0 + ty + i) * R + r0 + tx] = tile[tx][ty + i];
}

// ---------------------------------------------------------------- generic bf16 GEMM
// C[M,N] = A[M,K] * Bt[N,K]^T + bias[N]   (A,Bt bf16; C f32)
// Double-buffered: async DMA of stage s+1 overlaps WMMA of stage s; 1 barrier/stage.
#define BM 128
#define BN 128
#define BKG 64
#define LDT 72
__global__ __launch_bounds__(256) void gemm_bf16(const u16* __restrict__ A,
                                                 const u16* __restrict__ Bt,
                                                 const float* __restrict__ bias,
                                                 float* __restrict__ C,
                                                 int M, int N, int K) {
    __shared__ __align__(16) u16 As[2][BM * LDT];
    __shared__ __align__(16) u16 Bs[2][BN * LDT];
    int tid = threadIdx.x, lane = tid & 31, wid = tid >> 5;
    int bm = blockIdx.y * BM, bn = blockIdx.x * BN;
    int wm = (wid & 3) * 32;    // 4 waves along M
    int wn = (wid >> 2) * 64;   // 2 waves along N

    f32x8 acc[2][4];
#pragma unroll
    for (int a = 0; a < 2; ++a)
#pragma unroll
        for (int b = 0; b < 4; ++b) acc[a][b] = (f32x8){};

    // this thread's 4 staging chunks (16B each per tile)
    int r0c = (tid + 0)   >> 3, k0c = ((tid + 0)   & 7) * 8;
    int r1c = (tid + 256) >> 3, k1c = ((tid + 256) & 7) * 8;
    int r2c = (tid + 512) >> 3, k2c = ((tid + 512) & 7) * 8;
    int r3c = (tid + 768) >> 3, k3c = ((tid + 768) & 7) * 8;

    auto issue_stage = [&](int k0, int buf) {
        u16* as = As[buf];
        u16* bs = Bs[buf];
        async_cp16(&as[r0c * LDT + k0c], &A[(size_t)(bm + r0c) * K + k0 + k0c]);
        async_cp16(&bs[r0c * LDT + k0c], &Bt[(size_t)(bn + r0c) * K + k0 + k0c]);
        async_cp16(&as[r1c * LDT + k1c], &A[(size_t)(bm + r1c) * K + k0 + k1c]);
        async_cp16(&bs[r1c * LDT + k1c], &Bt[(size_t)(bn + r1c) * K + k0 + k1c]);
        async_cp16(&as[r2c * LDT + k2c], &A[(size_t)(bm + r2c) * K + k0 + k2c]);
        async_cp16(&bs[r2c * LDT + k2c], &Bt[(size_t)(bn + r2c) * K + k0 + k2c]);
        async_cp16(&as[r3c * LDT + k3c], &A[(size_t)(bm + r3c) * K + k0 + k3c]);
        async_cp16(&bs[r3c * LDT + k3c], &Bt[(size_t)(bn + r3c) * K + k0 + k3c]);
    };

    int nst = K / BKG;
    issue_stage(0, 0);
    for (int s = 0; s < nst; ++s) {
        wait_async0();          // stage s resident in LDS
        __syncthreads();
        if (s + 1 < nst) issue_stage((s + 1) * BKG, (s + 1) & 1);  // prefetch next
        const u16* as = As[s & 1];
        const u16* bs = Bs[s & 1];
#pragma unroll
        for (int kk = 0; kk < BKG; kk += 32) {
            bf16x16 a0 = frag_a(&as[(wm + 0)  * LDT + kk], LDT, lane);
            bf16x16 a1 = frag_a(&as[(wm + 16) * LDT + kk], LDT, lane);
#pragma unroll
            for (int j = 0; j < 4; ++j) {
                bf16x16 bfr = frag_b(&bs[(wn + 16 * j) * LDT + kk], LDT, lane);
                acc[0][j] = wmma_bf16(a0, bfr, acc[0][j]);
                acc[1][j] = wmma_bf16(a1, bfr, acc[1][j]);
            }
        }
    }
    int hi = lane >> 4, cl = lane & 15;
#pragma unroll
    for (int mi = 0; mi < 2; ++mi)
#pragma unroll
        for (int j = 0; j < 4; ++j) {
            int col = bn + wn + 16 * j + cl;
            float bv = bias[col];
#pragma unroll
            for (int p = 0; p < 8; ++p) {
                int row = bm + wm + 16 * mi + p + 8 * hi;
                C[(size_t)row * N + col] = acc[mi][j][p] + bv;
            }
        }
}

// ---------------------------------------------------------------- RoPE + head reorg
// qkv: [B*T, 3072] f32 -> Q,K: [B*H, T, 64] bf16 (Q pre-scaled by 1/8), VT: [B*H, 64, T] bf16
__global__ void rope_reorg(const float* __restrict__ qkv, u16* __restrict__ Qo,
                           u16* __restrict__ Ko, u16* __restrict__ VTo, int T) {
    long long idx = (long long)blockIdx.x * 256 + threadIdx.x;  // B*T*16*32
    int i = (int)(idx & 31);          // rotation pair index
    int h = (int)((idx >> 5) & 15);
    long long bt = idx >> 9;          // b*T + t
    int t = (int)(bt % T);
    long long b = bt / T;
    const float* base = qkv + bt * 3072 + h * 64 + 2 * i;
    float qe = base[0],    qo = base[1];
    float ke = base[1024], ko = base[1025];
    float ve = base[2048], vo = base[2049];
    float inv_freq = __expf(-(float)(2 * i) * (9.210340371976184f / 64.0f)); // ln(1e4)/64
    float sn, cs;
    sincosf((float)t * inv_freq, &sn, &cs);
    long long bh = b * 16 + h;
    const float qs = 0.125f;  // 1/sqrt(64) folded into Q
    u16 q0 = f32_to_bf16((qe * cs - qo * sn) * qs);
    u16 q1 = f32_to_bf16((qe * sn + qo * cs) * qs);
    u16 k0 = f32_to_bf16(ke * cs - ko * sn);
    u16 k1 = f32_to_bf16(ke * sn + ko * cs);
    *(unsigned*)&Qo[(bh * T + t) * 64 + 2 * i] = (unsigned)q0 | ((unsigned)q1 << 16);
    *(unsigned*)&Ko[(bh * T + t) * 64 + 2 * i] = (unsigned)k0 | ((unsigned)k1 << 16);
    VTo[(bh * 64 + 2 * i)     * T + t] = f32_to_bf16(ve);
    VTo[(bh * 64 + 2 * i + 1) * T + t] = f32_to_bf16(vo);
}

// per-batch "all keys padded" flag
__global__ void allpad_kernel(const int* __restrict__ pad, int* __restrict__ allpad, int T) {
    __shared__ int red[256];
    int b = blockIdx.x, t = threadIdx.x;
    int v = 1;
    for (int i = t; i < T; i += 256) v &= (pad[b * T + i] != 0);
    red[t] = v;
    __syncthreads();
    for (int s = 128; s > 0; s >>= 1) {
        if (t < s) red[t] &= red[t + s];
        __syncthreads();
    }
    if (t == 0) allpad[b] = red[0];
}

// ---------------------------------------------------------------- flash attention
// grid: (B*H, T/128); block 256 = 8 waves; wave owns a 16-query tile.
// Double-buffered K/V chunks staged via async global->LDS DMA.
__global__ __launch_bounds__(256) void attn_kernel(const u16* __restrict__ Q,
                                                   const u16* __restrict__ Kd,
                                                   const u16* __restrict__ VT,
                                                   const int* __restrict__ pad,
                                                   const int* __restrict__ allpad,
                                                   u16* __restrict__ Out, int T) {
    __shared__ __align__(16) u16 Ks[2][64 * LDT];
    __shared__ __align__(16) u16 Vs[2][64 * LDT];
    __shared__ __align__(16) u16 Ps[8][16 * LDT];
    int tid = threadIdx.x, lane = tid & 31, wid = tid >> 5;
    int bh = blockIdx.x, b = bh >> 4, h = bh & 15;
    int t0 = blockIdx.y * 128 + wid * 16;
    int r = lane & 15, hi = lane >> 4;

    const u16* Qh = Q  + (size_t)bh * T * 64;
    const u16* Kh = Kd + (size_t)bh * T * 64;
    const u16* Vh = VT + (size_t)bh * 64 * T;

    // Q fragments for K=0..31 and K=32..63 (held for the whole key loop)
    FragBF qf0, qf1;
    {
        const u16* qrow = Qh + (size_t)(t0 + r) * 64;
        qf0.h[0] = *(const u16x8*)(qrow + hi * 8);
        qf0.h[1] = *(const u16x8*)(qrow + 16 + hi * 8);
        qf1.h[0] = *(const u16x8*)(qrow + 32 + hi * 8);
        qf1.h[1] = *(const u16x8*)(qrow + 48 + hi * 8);
    }
    float mrow[8], lrow[8];
#pragma unroll
    for (int p = 0; p < 8; ++p) { mrow[p] = NEGINF; lrow[p] = 0.f; }
    f32x8 oacc[4];
#pragma unroll
    for (int j = 0; j < 4; ++j) oacc[j] = (f32x8){};
    int ap = allpad[b];

    // this thread's 2 staging chunks per tile (16B each)
    int r0c = (tid + 0)   >> 3, k0c = ((tid + 0)   & 7) * 8;
    int r1c = (tid + 256) >> 3, k1c = ((tid + 256) & 7) * 8;
    auto issue_chunk = [&](int kc, int buf) {
        async_cp16(&Ks[buf][r0c * LDT + k0c], &Kh[(size_t)(kc + r0c) * 64 + k0c]);
        async_cp16(&Vs[buf][r0c * LDT + k0c], &Vh[(size_t)r0c * T + kc + k0c]);
        async_cp16(&Ks[buf][r1c * LDT + k1c], &Kh[(size_t)(kc + r1c) * 64 + k1c]);
        async_cp16(&Vs[buf][r1c * LDT + k1c], &Vh[(size_t)r1c * T + kc + k1c]);
    };

    int nch = T / 64;
    issue_chunk(0, 0);
    for (int ci = 0; ci < nch; ++ci) {
        wait_async0();          // chunk ci resident
        __syncthreads();
        if (ci + 1 < nch) issue_chunk((ci + 1) * 64, (ci + 1) & 1);  // prefetch next
        const u16* ks = Ks[ci & 1];
        const u16* vs = Vs[ci & 1];
        int kc = ci * 64;

        // ---- scores S[16,64] = Qtile @ K^T (Q already scaled by 1/sqrt(d))
        f32x8 s[4];
#pragma unroll
        for (int j = 0; j < 4; ++j) {
            s[j] = (f32x8){};
            bf16x16 b0 = frag_b(&ks[(16 * j) * LDT + 0],  LDT, lane);
            bf16x16 b1 = frag_b(&ks[(16 * j) * LDT + 32], LDT, lane);
            s[j] = wmma_bf16(qf0.bf, b0, s[j]);
            s[j] = wmma_bf16(qf1.bf, b1, s[j]);
        }

        // ---- online softmax (lane owns column kc+16j+(lane&15), rows p+8*hi)
        float vmul[4], sv[4][8], cmax[8];
#pragma unroll
        for (int p = 0; p < 8; ++p) cmax[p] = NEGINF;
#pragma unroll
        for (int j = 0; j < 4; ++j) {
            int kg = kc + 16 * j + r;
            vmul[j] = (ap | (pad[b * T + kg] == 0)) ? 1.f : 0.f;
#pragma unroll
            for (int p = 0; p < 8; ++p) {
                float x = (vmul[j] != 0.f) ? s[j][p] : NEGINF;
                sv[j][p] = x;
                cmax[p] = fmaxf(cmax[p], x);
            }
        }
#pragma unroll
        for (int p = 0; p < 8; ++p) {             // row-max across the 16 lanes of a half
            for (int d = 8; d >= 1; d >>= 1)
                cmax[p] = fmaxf(cmax[p], __shfl_xor(cmax[p], d));
        }
        float alpha[8], rsum[8];
#pragma unroll
        for (int p = 0; p < 8; ++p) {
            float mnew = fmaxf(mrow[p], cmax[p]);
            alpha[p] = exp2f((mrow[p] - mnew) * LOG2E);
            mrow[p] = mnew;
            rsum[p] = 0.f;
        }
#pragma unroll
        for (int j = 0; j < 4; ++j)
#pragma unroll
            for (int p = 0; p < 8; ++p) {
                float pv = exp2f((sv[j][p] - mrow[p]) * LOG2E) * vmul[j];
                sv[j][p] = pv;
                rsum[p] += pv;
            }
#pragma unroll
        for (int p = 0; p < 8; ++p) {
            for (int d = 8; d >= 1; d >>= 1) rsum[p] += __shfl_xor(rsum[p], d);
            lrow[p] = lrow[p] * alpha[p] + rsum[p];
        }
#pragma unroll
        for (int j = 0; j < 4; ++j)
#pragma unroll
            for (int p = 0; p < 8; ++p) oacc[j][p] *= alpha[p];

        // ---- P -> per-wave LDS (C-layout -> row-major bf16), then O += P @ V
#pragma unroll
        for (int j = 0; j < 4; ++j)
#pragma unroll
            for (int p = 0; p < 8; ++p)
                Ps[wid][(p + 8 * hi) * LDT + 16 * j + r] = f32_to_bf16(sv[j][p]);
        // same-wave LDS ops are in-order; no block barrier needed for Ps
#pragma unroll
        for (int kk = 0; kk < 64; kk += 32) {
            bf16x16 pa = frag_a(&Ps[wid][kk], LDT, lane);
#pragma unroll
            for (int j = 0; j < 4; ++j) {
                bf16x16 vb = frag_b(&vs[(16 * j) * LDT + kk], LDT, lane);
                oacc[j] = wmma_bf16(pa, vb, oacc[j]);
            }
        }
    }
    // ---- normalize and store [B*T, H*64] bf16
#pragma unroll
    for (int p = 0; p < 8; ++p) {
        float inv = 1.0f / lrow[p];
        int row = b * T + t0 + p + 8 * hi;
#pragma unroll
        for (int j = 0; j < 4; ++j)
            Out[(size_t)row * 1024 + h * 64 + 16 * j + r] = f32_to_bf16(oacc[j][p] * inv);
    }
}

// ---------------------------------------------------------------- launcher
extern "C" void kernel_launch(void* const* d_in, const int* in_sizes, int n_in,
                              void* d_out, int out_size, void* d_ws, size_t ws_size,
                              hipStream_t stream) {
    const float* tokens = (const float*)d_in[0];
    const int*   pad    = (const int*)  d_in[1];
    const float* W_qkv  = (const float*)d_in[2];
    const float* b_qkv  = (const float*)d_in[3];
    const float* W_out  = (const float*)d_in[4];
    const float* b_out  = (const float*)d_in[5];
    float* out = (float*)d_out;

    const int B = 4, T = 2048, D = 1024, BT = B * T;

    size_t off = 0;
    auto carve = [&](size_t bytes) {
        void* p = (char*)d_ws + off;
        off += (bytes + 255) & ~(size_t)255;
        return p;
    };
    u16*  Xb  = (u16*)  carve((size_t)BT * D * 2);        // tokens bf16
    u16*  WqT = (u16*)  carve((size_t)3 * D * D * 2);     // W_qkv^T bf16 [3072,1024]
    u16*  WoT = (u16*)  carve((size_t)D * D * 2);         // W_out^T bf16
    float* qkv = (float*)carve((size_t)BT * 3 * D * 4);   // qkv f32
    u16*  Qb  = (u16*)  carve((size_t)BT * D * 2);        // [B*H, T, 64]
    u16*  Kb  = (u16*)  carve((size_t)BT * D * 2);
    u16*  VTb = (u16*)  carve((size_t)BT * D * 2);        // [B*H, 64, T]
    u16*  AOb = (u16*)  carve((size_t)BT * D * 2);        // attention output bf16
    int*  apd = (int*)  carve(64);

    int nTok = BT * D;
    f32_to_bf16_kernel<<<(nTok + 255) / 256, 256, 0, stream>>>(tokens, Xb, nTok);
    transpose_f32_bf16<<<dim3(3 * D / 32, D / 32), 256, 0, stream>>>(W_qkv, WqT, D, 3 * D);
    transpose_f32_bf16<<<dim3(D / 32, D / 32), 256, 0, stream>>>(W_out, WoT, D, D);

    gemm_bf16<<<dim3(3 * D / BN, BT / BM), 256, 0, stream>>>(Xb, WqT, b_qkv, qkv,
                                                             BT, 3 * D, D);

    rope_reorg<<<(B * T * 16 * 32) / 256, 256, 0, stream>>>(qkv, Qb, Kb, VTb, T);
    allpad_kernel<<<B, 256, 0, stream>>>(pad, apd, T);

    attn_kernel<<<dim3(B * 16, T / 128), 256, 0, stream>>>(Qb, Kb, VTb, pad, apd, AOb, T);

    gemm_bf16<<<dim3(D / BN, BT / BM), 256, 0, stream>>>(AOb, WoT, b_out, out,
                                                         BT, D, D);
}
